// HQQLinearLoRA_75531294867844
// MI455X (gfx1250) — compile-verified, matrix operand order checked by reading, not run on hardware
//
#include <hip/hip_runtime.h>
#include <stdint.h>

// ---------------------------------------------------------------------------
// out = x @ (W + lora_A @ lora_B) + bias         (SCALING == 1.0, exact fold)
//
// bf16x3 split-precision GEMM on v_wmma_f32_16x16x32_bf16 (fp32 accumulate):
//   C += Ahi*Bhi + Ahi*Blo + Alo*Bhi
//
// Workspace: Wt_hi / Wt_lo = bf16 planes of (W + A@B), stored TRANSPOSED
// ([n][k]) so GEMM B-fragments are contiguous-K ds_load_b128s. 64 MB total.
// LDS tiles are stored in exact WMMA fragment order -> zero unpack VALU.
// W' tiles are staged with GLOBAL_LOAD_ASYNC_TO_LDS_B128 (ASYNCcnt path).
// ---------------------------------------------------------------------------

typedef __attribute__((ext_vector_type(16))) __bf16        v16bf;
typedef __attribute__((ext_vector_type(8)))  float         v8f;

#define D_DIM 4096
#define M_TOT 16384
#define R_DIM 16
#define BM 128
#define BN 128
#define BK 32
#define NKT (D_DIM / BK)      // 128 K-steps
#define ROW_DW 20             // 16 data dwords (32 bf16) + 4 pad -> 80B stride

struct Frag { uint4 a, b; };

// Load one 16x32 bf16 fragment (A or B side) from an LDS row laid out as
// 32 bf16 in K order. halfSel = lane>=16. Two ds_load_b128, no VALU.
__device__ __forceinline__ v16bf load_frag(const uint32_t* row, uint32_t halfSel) {
  Frag f;
  f.a = *(const uint4*)(row + halfSel * 4u);        // K k0..k0+7
  f.b = *(const uint4*)(row + 8u + halfSel * 4u);   // K k0+16..k0+23
  return __builtin_bit_cast(v16bf, f);
}

// pack high 16 bits of u0 (low half) and u1 (high half) into one dword
__device__ __forceinline__ uint32_t pack_hi16(uint32_t u1, uint32_t u0) {
  return __builtin_amdgcn_perm(u1, u0, 0x07060302u);
}

// async 16-byte global -> LDS copy (per-lane addresses), ASYNCcnt-tracked
__device__ __forceinline__ void async_g2l_b128(const void* g, void* lds) {
  uint32_t loff = (uint32_t)(uintptr_t)lds;   // LDS aperture低32 = LDS offset
  asm volatile("global_load_async_to_lds_b128 %0, %1, off"
               :: "v"(loff), "v"((uint64_t)(uintptr_t)g)
               : "memory");
}

__device__ __forceinline__ void wait_async0() {
  asm volatile("s_wait_asynccnt 0x0" ::: "memory");
}

// ---------------------------------------------------------------------------
// Kernel 1: W' = W + A@B ; write bf16 hi/lo planes TRANSPOSED: Wt[n][k].
// 64x64 tiles, LDS transpose, coalesced in and 32B-chunk out.
// ---------------------------------------------------------------------------
__global__ __launch_bounds__(256) void fold_w_kernel(
    const float* __restrict__ W, const float* __restrict__ lA,
    const float* __restrict__ lB, uint16_t* __restrict__ Wt_hi,
    uint16_t* __restrict__ Wt_lo) {
  __shared__ uint16_t sh[64][72];
  __shared__ uint16_t sl[64][72];
  const uint32_t t  = threadIdx.x;
  const uint32_t k0 = blockIdx.y * 64u;
  const uint32_t n0 = blockIdx.x * 64u;

#pragma unroll
  for (int i = 0; i < 4; ++i) {
    uint32_t v = t + i * 256u;
    uint32_t r  = v >> 4;      // tile row (k)
    uint32_t c4 = v & 15u;     // float4 within row (n)
    float4 wv = *(const float4*)(W + (size_t)(k0 + r) * D_DIM + n0 + c4 * 4u);
    float a[4] = {wv.x, wv.y, wv.z, wv.w};
#pragma unroll
    for (int rr = 0; rr < R_DIM; ++rr) {
      float av = lA[(size_t)(k0 + r) * R_DIM + rr];
      const float* bp = lB + (size_t)rr * D_DIM + n0 + c4 * 4u;
#pragma unroll
      for (int j = 0; j < 4; ++j) a[j] = fmaf(av, bp[j], a[j]);
    }
#pragma unroll
    for (int j = 0; j < 4; ++j) {
      uint32_t u = __float_as_uint(a[j]);
      float lo = a[j] - __uint_as_float(u & 0xffff0000u);   // exact residual
      sh[r][c4 * 4u + j] = (uint16_t)(u >> 16);
      sl[r][c4 * 4u + j] = (uint16_t)(__float_as_uint(lo) >> 16);
    }
  }
  __syncthreads();

  // transposed write: 16 consecutive k per thread for one n
  const uint32_t n  = t >> 2;
  const uint32_t kc = (t & 3u) * 16u;
  uint32_t dh[8], dl[8];
#pragma unroll
  for (int j = 0; j < 8; ++j) {
    dh[j] = (uint32_t)sh[kc + 2 * j][n] | ((uint32_t)sh[kc + 2 * j + 1][n] << 16);
    dl[j] = (uint32_t)sl[kc + 2 * j][n] | ((uint32_t)sl[kc + 2 * j + 1][n] << 16);
  }
  const size_t obase = (size_t)(n0 + n) * D_DIM + k0 + kc;
  *(uint4*)(Wt_hi + obase)      = make_uint4(dh[0], dh[1], dh[2], dh[3]);
  *(uint4*)(Wt_hi + obase + 8)  = make_uint4(dh[4], dh[5], dh[6], dh[7]);
  *(uint4*)(Wt_lo + obase)      = make_uint4(dl[0], dl[1], dl[2], dl[3]);
  *(uint4*)(Wt_lo + obase + 8)  = make_uint4(dl[4], dl[5], dl[6], dl[7]);
}

// ---------------------------------------------------------------------------
// Kernel 2: Out = X @ W'(hi/lo) + bias   via bf16x3 WMMA
//   grid = (N/128, M/128), 256 threads = 8 waves, wave patch = 64x32
// ---------------------------------------------------------------------------
__global__ __launch_bounds__(256) void lora_gemm_kernel(
    const float* __restrict__ X, const uint16_t* __restrict__ Wt_hi,
    const uint16_t* __restrict__ Wt_lo, const float* __restrict__ bias,
    float* __restrict__ Out) {
  // [buf][plane(hi/lo)][row][ROW_DW dwords]
  __shared__ uint32_t As[2][2][BM][ROW_DW];   // x tile,  bf16 pairs, K-major
  __shared__ uint32_t Bs[2][2][BN][ROW_DW];   // W' tile, bf16 pairs, K-major

  const uint32_t tid  = threadIdx.x;
  const uint32_t lane = tid & 31u;
  const uint32_t wave = tid >> 5;
  const uint32_t hsel = lane >> 4;     // WMMA K-half select
  const uint32_t l16  = lane & 15u;
  const uint32_t wm   = wave & 1u;     // 2 waves down   -> 64 rows each
  const uint32_t wn   = wave >> 1;     // 4 waves across -> 32 cols each

  const uint32_t bn0 = blockIdx.x * BN;
  const uint32_t bm0 = blockIdx.y * BM;
  const uint16_t* Wt[2] = {Wt_hi, Wt_lo};

  v8f acc[4][2];
#pragma unroll
  for (int mi = 0; mi < 4; ++mi)
#pragma unroll
    for (int ni = 0; ni < 2; ++ni)
      acc[mi][ni] = (v8f){0.f, 0.f, 0.f, 0.f, 0.f, 0.f, 0.f, 0.f};

  // ---------------- staging helpers (inlined twice) ----------------
  // A: thread i-th float4: v = tid + i*256 ; m = v>>3 ; kc = v&7 (4 fp32)
  // B: thread i-th b128 per plane: v = tid + i*256 ; col = v>>2 ; c16 = v&3

#define STAGE_A(BUF, KT)                                                      \
  {                                                                           \
    const float* xbase = X + (size_t)bm0 * D_DIM + (size_t)(KT) * BK;         \
    _Pragma("unroll") for (int i = 0; i < 4; ++i) {                           \
      uint32_t v = tid + i * 256u;                                            \
      uint32_t m = v >> 3, kc = v & 7u;                                       \
      float4 xv = *(const float4*)(xbase + (size_t)m * D_DIM + kc * 4u);      \
      uint32_t u0 = __float_as_uint(xv.x), u1 = __float_as_uint(xv.y);        \
      uint32_t u2 = __float_as_uint(xv.z), u3 = __float_as_uint(xv.w);        \
      float l0 = xv.x - __uint_as_float(u0 & 0xffff0000u);                    \
      float l1 = xv.y - __uint_as_float(u1 & 0xffff0000u);                    \
      float l2 = xv.z - __uint_as_float(u2 & 0xffff0000u);                    \
      float l3 = xv.w - __uint_as_float(u3 & 0xffff0000u);                    \
      uint2 hp = make_uint2(pack_hi16(u1, u0), pack_hi16(u3, u2));            \
      uint2 lp = make_uint2(                                                  \
          pack_hi16(__float_as_uint(l1), __float_as_uint(l0)),                \
          pack_hi16(__float_as_uint(l3), __float_as_uint(l2)));               \
      *(uint2*)&As[BUF][0][m][kc * 2u] = hp;                                  \
      *(uint2*)&As[BUF][1][m][kc * 2u] = lp;                                  \
    }                                                                         \
  }

#define STAGE_B(BUF, KT)                                                      \
  {                                                                           \
    _Pragma("unroll") for (int p = 0; p < 2; ++p) {                           \
      const uint16_t* wb = Wt[p] + (size_t)bn0 * D_DIM + (size_t)(KT) * BK;   \
      _Pragma("unroll") for (int i = 0; i < 2; ++i) {                         \
        uint32_t v = tid + i * 256u;                                          \
        uint32_t col = v >> 2, c16 = v & 3u;                                  \
        async_g2l_b128(wb + (size_t)col * D_DIM + c16 * 8u,                   \
                       &Bs[BUF][p][col][c16 * 4u]);                           \
      }                                                                       \
    }                                                                         \
  }

  STAGE_A(0, 0)
  STAGE_B(0, 0)
  wait_async0();
  __syncthreads();

  for (int kt = 0; kt < NKT; ++kt) {
    const int cur = kt & 1;

    // ---- B fragments: 2 col tiles x {hi,lo}, pure ds_load_b128 ----
    v16bf bHi[2], bLo[2];
#pragma unroll
    for (int ni = 0; ni < 2; ++ni) {
      const uint32_t col = wn * 32u + ni * 16u + l16;
      bHi[ni] = load_frag(&Bs[cur][0][col][0], hsel);
      bLo[ni] = load_frag(&Bs[cur][1][col][0], hsel);
    }

    // ---- A fragments + 3x WMMA per tile ----
#pragma unroll
    for (int mi = 0; mi < 4; ++mi) {
      const uint32_t row = wm * 64u + mi * 16u + l16;
      v16bf aHi = load_frag(&As[cur][0][row][0], hsel);
      v16bf aLo = load_frag(&As[cur][1][row][0], hsel);
#pragma unroll
      for (int ni = 0; ni < 2; ++ni) {
        acc[mi][ni] = __builtin_amdgcn_wmma_f32_16x16x32_bf16(
            false, aHi, false, bHi[ni], (short)0, acc[mi][ni], false, false);
        acc[mi][ni] = __builtin_amdgcn_wmma_f32_16x16x32_bf16(
            false, aHi, false, bLo[ni], (short)0, acc[mi][ni], false, false);
        acc[mi][ni] = __builtin_amdgcn_wmma_f32_16x16x32_bf16(
            false, aLo, false, bHi[ni], (short)0, acc[mi][ni], false, false);
      }
    }

    // ---- stage next K-tile into the other buffer ----
    if (kt + 1 < NKT) {
      const int nb = cur ^ 1;
      STAGE_A(nb, kt + 1)
      STAGE_B(nb, kt + 1)
      wait_async0();
    }
    __syncthreads();
  }

  // ---- epilogue: add bias, store (C layout: M = r + 8*hsel, N = l16) ----
#pragma unroll
  for (int ni = 0; ni < 2; ++ni) {
    const uint32_t n = bn0 + wn * 32u + ni * 16u + l16;
    const float bv = bias[n];
#pragma unroll
    for (int mi = 0; mi < 4; ++mi) {
      const uint32_t mbase = bm0 + wm * 64u + mi * 16u + hsel * 8u;
      float* op = Out + (size_t)mbase * D_DIM + n;
#pragma unroll
      for (int r = 0; r < 8; ++r)
        op[(size_t)r * D_DIM] = acc[mi][ni][r] + bv;
    }
  }
#undef STAGE_A
#undef STAGE_B
}

// ---------------------------------------------------------------------------
extern "C" void kernel_launch(void* const* d_in, const int* in_sizes, int n_in,
                              void* d_out, int out_size, void* d_ws, size_t ws_size,
                              hipStream_t stream) {
  const float* x      = (const float*)d_in[0];   // [4,4096,4096]
  const float* W      = (const float*)d_in[1];   // [4096,4096]
  const float* bias   = (const float*)d_in[2];   // [4096]
  const float* lora_A = (const float*)d_in[3];   // [4096,16]
  const float* lora_B = (const float*)d_in[4];   // [16,4096]
  float* out          = (float*)d_out;           // [4,4096,4096]

  uint16_t* Wt_hi = (uint16_t*)d_ws;                       // 32 MB
  uint16_t* Wt_lo = Wt_hi + (size_t)D_DIM * D_DIM;         // 32 MB

  dim3 fgrid(D_DIM / 64, D_DIM / 64);
  fold_w_kernel<<<fgrid, 256, 0, stream>>>(W, lora_A, lora_B, Wt_hi, Wt_lo);

  dim3 grid(D_DIM / BN, M_TOT / BM);
  lora_gemm_kernel<<<grid, 256, 0, stream>>>(x, Wt_hi, Wt_lo, bias, out);
}
